// CustomLoss_30940944401001
// MI455X (gfx1250) — matrix operand order repaired and checked by reading, not compiled, not run on hardware
//
#include <hip/hip_runtime.h>
#include <hip/hip_bf16.h>

// ---------------------------------------------------------------------------
// CustomLoss on MI455X (gfx1250):
//   P = pred @ Xi            (8192x1024 @ 1024x4096, fp32 in, f32 accum)
//   gm = P[seg,yh] - P[seg,yt]; loss = sum(max(1-gm,0)) / 8192
//
// Pass 1: one-shot RNE f32->bf16 conversion (pred -> predB row-major,
//         Xi -> XiBt transposed) so the GEMM hot loop does zero conversion.
// Pass 2: bf16 WMMA GEMM (v_wmma_f32_16x16x32_bf16).
//         Block tile 256x128, 8 waves, 64x64 per wave (16 accumulators) ->
//         1 ds_load_b128 per WMMA. Double-buffered LDS filled with
//         global_load_async_to_lds_b128 (ASYNCcnt DMA), s_wait_asynccnt+barrier.
// Pass 3: gather + hinge + deterministic tree reduction (P is L2-resident:
//         128 MB < 192 MB L2, so the random gather mostly hits L2).
//
// Workspace layout (needs ~152 MB):
//   [0,128MB)            P            f32 8192x4096
//   [128MB, +2KB)        partials     512 f32
//   [128MB+4KB, +16MB)   predB        bf16 8192x1024
//   [.. , +8MB)          XiBt         bf16 4096x1024 (= Xi^T)
// ---------------------------------------------------------------------------

typedef __attribute__((ext_vector_type(16))) __bf16   v16bf;
typedef __attribute__((ext_vector_type(8)))  float    v8f;
typedef __attribute__((ext_vector_type(4)))  float    f4v;
typedef __attribute__((ext_vector_type(4)))  unsigned u4v;

#define M_DIM 8192
#define D_DIM 1024   // reduction dim
#define K_DIM 4096   // output cols
#define TOTAL_L 1048576

#define BM 256
#define BN 128
#define BK 32
#define NK (D_DIM / BK)
#define LDS_STRIDE 80              // 32 bf16 = 64B data + 16B pad (conflict-free b128 frag loads)
#define A_BYTES (BM * LDS_STRIDE)  // 20480
#define B_BYTES (BN * LDS_STRIDE)  // 10240
#define BUF_BYTES (A_BYTES + B_BYTES)  // 30720 (x2 buffers = 60 KB LDS)

// round-to-nearest-even f32 -> bf16, packed pair
__device__ __forceinline__ unsigned pk_bf16(float lo, float hi) {
  unsigned a = __float_as_uint(lo);
  unsigned b = __float_as_uint(hi);
  a += 0x7FFFu + ((a >> 16) & 1u);
  b += 0x7FFFu + ((b >> 16) & 1u);
  return (a >> 16) | (b & 0xFFFF0000u);
}

union Frag {      // 32 bytes: 16 bf16 elements of a WMMA A/B operand
  u4v   q[2];
  v16bf v;
};

// ---------------- pass 1a: straight f32 -> bf16 (pred) ----------------------
__global__ __launch_bounds__(256) void convert_bf16(
    const float* __restrict__ src, unsigned short* __restrict__ dst) {
  const size_t base = ((size_t)blockIdx.x * 256 + threadIdx.x) * 16;
  f4v a0 = *(const f4v*)(src + base);
  f4v a1 = *(const f4v*)(src + base + 4);
  f4v a2 = *(const f4v*)(src + base + 8);
  f4v a3 = *(const f4v*)(src + base + 12);
  u4v w0 = {pk_bf16(a0.x, a0.y), pk_bf16(a0.z, a0.w),
            pk_bf16(a1.x, a1.y), pk_bf16(a1.z, a1.w)};
  u4v w1 = {pk_bf16(a2.x, a2.y), pk_bf16(a2.z, a2.w),
            pk_bf16(a3.x, a3.y), pk_bf16(a3.z, a3.w)};
  *(u4v*)(dst + base)     = w0;
  *(u4v*)(dst + base + 8) = w1;
}

// ------------- pass 1b: f32 -> bf16 with transpose (Xi -> Xi^T) -------------
// Reads coalesced along n (lanes = consecutive n), writes 32B runs along k.
__global__ __launch_bounds__(256) void convert_xiT(
    const float* __restrict__ Xi, unsigned short* __restrict__ XiBt) {
  const int n   = blockIdx.x * 256 + threadIdx.x;  // 0..4095
  const int kr0 = blockIdx.y * 16;                 // 0..1008
  float f[16];
#pragma unroll
  for (int i = 0; i < 16; ++i) f[i] = Xi[(size_t)(kr0 + i) * K_DIM + n];
  u4v w0 = {pk_bf16(f[0], f[1]),  pk_bf16(f[2], f[3]),
            pk_bf16(f[4], f[5]),  pk_bf16(f[6], f[7])};
  u4v w1 = {pk_bf16(f[8], f[9]),  pk_bf16(f[10], f[11]),
            pk_bf16(f[12], f[13]), pk_bf16(f[14], f[15])};
  *(u4v*)(XiBt + (size_t)n * D_DIM + kr0)     = w0;
  *(u4v*)(XiBt + (size_t)n * D_DIM + kr0 + 8) = w1;
}

// ---------------- pass 2: bf16 WMMA GEMM with async LDS fill ----------------
__device__ __forceinline__ void wait_async0() {
  asm volatile("s_wait_asynccnt 0x0" ::: "memory");
}

__device__ __forceinline__ void async_b128(unsigned lds_addr, const void* g) {
  asm volatile("global_load_async_to_lds_b128 %0, %1, off"
               :: "v"(lds_addr), "v"((unsigned long long)(uintptr_t)g)
               : "memory");
}

__global__ __launch_bounds__(256) void gemm_bf16_wmma(
    const unsigned short* __restrict__ predB,
    const unsigned short* __restrict__ XiBt,
    float* __restrict__ P) {
  __shared__ u4v lds_backing[2 * BUF_BYTES / 16];
  char* lds = (char*)lds_backing;

  const int tid  = threadIdx.x;
  const int lane = tid & 31;
  const int wave = tid >> 5;
  const int mBase = blockIdx.y * BM;
  const int nBase = blockIdx.x * BN;
  const int waveM = wave >> 1;  // 0..3  -> 64-row slice of 256
  const int waveN = wave & 1;   // 0..1  -> 64-col slice of 128

  // Staging: all 256 threads DMA 6 x b128 per k-step.
  //   A: thread t copies all 4 chunks of A row t (64B).
  //   B: thread t copies 2 chunks of B row t>>1 (chunks (t&1)*2 + {0,1}).
  const char* gsrcA = (const char*)(predB + (size_t)(mBase + tid) * D_DIM);
  const char* gsrcB =
      (const char*)(XiBt + (size_t)(nBase + (tid >> 1)) * D_DIM) + (tid & 1) * 32;
  const unsigned ldsA0 = (unsigned)(uintptr_t)lds + tid * LDS_STRIDE;
  const unsigned ldsB0 = (unsigned)(uintptr_t)lds + A_BYTES +
                         (tid >> 1) * LDS_STRIDE + (tid & 1) * 32;

  auto issue_async = [&](int ks, int buf) {
    const size_t koff = (size_t)ks * (BK * 2);  // bytes into the k dimension
    const unsigned lb = buf * BUF_BYTES;
#pragma unroll
    for (int c = 0; c < 4; ++c)
      async_b128(ldsA0 + lb + c * 16, gsrcA + koff + c * 16);
#pragma unroll
    for (int e = 0; e < 2; ++e)
      async_b128(ldsB0 + lb + e * 16, gsrcB + koff + e * 16);
  };

  v8f acc[4][4];
  const v8f vzero = {0.f, 0.f, 0.f, 0.f, 0.f, 0.f, 0.f, 0.f};
#pragma unroll
  for (int i = 0; i < 4; ++i)
#pragma unroll
    for (int j = 0; j < 4; ++j) acc[i][j] = vzero;

  issue_async(0, 0);
  wait_async0();
  __syncthreads();

  for (int ks = 0; ks < NK; ++ks) {
    const int buf = ks & 1;
    if (ks + 1 < NK) issue_async(ks + 1, buf ^ 1);   // DMA next buffer (no wait)

    const char* As = lds + buf * BUF_BYTES;
    const char* Bs = As + A_BYTES;

    // A fragment (16x32 bf16): lane m = L&15; K half = (L>>4)*16, contiguous
    Frag af[4], bfr[4];
    const int aoff = (lane >> 4) * 32;  // bytes
    const int arow = lane & 15;
#pragma unroll
    for (int i = 0; i < 4; ++i) {
      const char* pA = As + (waveM * 64 + i * 16 + arow) * LDS_STRIDE + aoff;
      af[i].q[0] = *(const u4v*)(pA);
      af[i].q[1] = *(const u4v*)(pA + 16);
    }
    // B fragment (32x16 bf16): lane n = L&15; K half = (L>>4)*16, contiguous
    const int boff = (lane >> 4) * 32;  // bytes
#pragma unroll
    for (int j = 0; j < 4; ++j) {
      const char* pB = Bs + (waveN * 64 + j * 16 + arow) * LDS_STRIDE + boff;
      bfr[j].q[0] = *(const u4v*)(pB);
      bfr[j].q[1] = *(const u4v*)(pB + 16);
    }

#pragma unroll
    for (int i = 0; i < 4; ++i)
#pragma unroll
      for (int j = 0; j < 4; ++j)
        acc[i][j] = __builtin_amdgcn_wmma_f32_16x16x32_bf16(
            false, af[i].v, false, bfr[j].v, (short)0, acc[i][j], false, false);

    wait_async0();     // next buffer fully in LDS (own wave's DMAs)
    __syncthreads();   // ... and everyone else's
  }

  // Epilogue: C/D layout — lane n = L&15, vgpr rr -> row rr + 8*(L>>4)
  const int mW = mBase + waveM * 64;
  const int nW = nBase + waveN * 64;
  const int rn = lane & 15;
  const int rhalf = (lane >> 4) * 8;
#pragma unroll
  for (int i = 0; i < 4; ++i)
#pragma unroll
    for (int j = 0; j < 4; ++j) {
      const int n = nW + j * 16 + rn;
#pragma unroll
      for (int rr = 0; rr < 8; ++rr) {
        const int m = mW + i * 16 + rr + rhalf;
        P[(size_t)m * K_DIM + n] = acc[i][j][rr];
      }
    }
}

// ------- pass 3: gather + hinge + per-block partials (deterministic) --------
__global__ __launch_bounds__(256) void gather_hinge(
    const float* __restrict__ P, const int* __restrict__ yh,
    const int* __restrict__ yt, const int* __restrict__ seg,
    float* __restrict__ partials) {
  __shared__ float red[256];
  const int tid = threadIdx.x;
  const size_t base = (size_t)blockIdx.x * 2048;
  float s = 0.f;
#pragma unroll
  for (int rr = 0; rr < 8; ++rr) {
    const size_t l = base + rr * 256 + tid;
    const size_t row = (size_t)seg[l] * K_DIM;
    const float gm = P[row + yh[l]] - P[row + yt[l]];
    s += fmaxf(1.0f - gm, 0.0f);
  }
  red[tid] = s;
  __syncthreads();
  for (int off = 128; off > 0; off >>= 1) {
    if (tid < off) red[tid] += red[tid + off];
    __syncthreads();
  }
  if (tid == 0) partials[blockIdx.x] = red[0];
}

__global__ __launch_bounds__(256) void finalize_loss(
    const float* __restrict__ partials, float* __restrict__ out) {
  __shared__ float red[256];
  const int tid = threadIdx.x;
  red[tid] = partials[tid] + partials[tid + 256];
  __syncthreads();
  for (int off = 128; off > 0; off >>= 1) {
    if (tid < off) red[tid] += red[tid + off];
    __syncthreads();
  }
  if (tid == 0) out[0] = red[0] / (float)M_DIM;
}

extern "C" void kernel_launch(void* const* d_in, const int* in_sizes, int n_in,
                              void* d_out, int out_size, void* d_ws, size_t ws_size,
                              hipStream_t stream) {
  const float* pred = (const float*)d_in[0];
  const float* Xi   = (const float*)d_in[1];
  const int*   yh   = (const int*)d_in[2];   // int64 in reference -> int32 per harness
  const int*   yt   = (const int*)d_in[3];
  const int*   seg  = (const int*)d_in[4];

  char* ws = (char*)d_ws;
  const size_t pBytes = (size_t)M_DIM * K_DIM * sizeof(float);  // 128 MB
  float*          P        = (float*)ws;
  float*          partials = (float*)(ws + pBytes);
  unsigned short* predB    = (unsigned short*)(ws + pBytes + 4096);
  unsigned short* XiBt     = predB + (size_t)M_DIM * D_DIM;

  // pass 1: one-shot bf16 conversion (+ transpose of Xi)
  convert_bf16<<<(M_DIM * D_DIM) / (256 * 16), 256, 0, stream>>>(pred, predB);
  convert_xiT<<<dim3(K_DIM / 256, D_DIM / 16), 256, 0, stream>>>(Xi, XiBt);

  // pass 2: WMMA GEMM
  dim3 grid(K_DIM / BN, M_DIM / BM);  // 32 x 32
  gemm_bf16_wmma<<<grid, 256, 0, stream>>>(predB, XiBt, P);

  // pass 3: loss
  gather_hinge<<<TOTAL_L / 2048, 256, 0, stream>>>(P, yh, yt, seg, partials);
  finalize_loss<<<1, 256, 0, stream>>>(partials, (float*)d_out);
}
